// DenseMotionNetwork_18116172055181
// MI455X (gfx1250) — compile-verified, matrix operand order checked by reading, not run on hardware
//
#include <hip/hip_runtime.h>
#include <hip/hip_bf16.h>

typedef __attribute__((ext_vector_type(16))) _Float16 v16h;
typedef __attribute__((ext_vector_type(8)))  _Float16 h8;
typedef __attribute__((ext_vector_type(8)))  float    v8f;

#define BS 8
#define IMG_H 128
#define IMG_W 128
#define NKP 10

// ---------------------------------------------------------------------------
// Weight repack: f32 (kh,kw,Ci,Co) -> f16 WMMA-B-fragment layout
//   dst[(((t*KB+kb)*Co_pad + n)*2 + half)*16 + e] = w[t][kb*32 + kmap(e,half)][n]
//   kmap(e,half) = (e<8 ? e : e+8) + half*8   (matches 16-bit A/B VGPR layout)
// ---------------------------------------------------------------------------
__global__ void convert_w(const float* __restrict__ w, int taps, int Ci_real,
                          int Ci_pad, int Co_real, int Co_pad,
                          _Float16* __restrict__ dst, int total)
{
    int idx = blockIdx.x * blockDim.x + threadIdx.x;
    if (idx >= total) return;
    int e    = idx & 15;
    int half = (idx >> 4) & 1;
    int rem  = idx >> 5;
    int n    = rem % Co_pad;
    rem /= Co_pad;
    int KB = Ci_pad >> 5;
    int kb = rem % KB;
    int t  = rem / KB;
    int k  = (kb << 5) + ((e < 8) ? e : e + 8) + half * 8;
    float v = 0.0f;
    if (k < Ci_real && n < Co_real)
        v = w[((size_t)t * Ci_real + k) * Co_real + n];
    dst[idx] = (_Float16)v;
}

// ---------------------------------------------------------------------------
// LDS-staged implicit-GEMM conv (3x3 or 7x7), SAME padding, optional fused
// x2 nearest upsample on input addressing, fused relu(gamma*(conv+b)+beta).
// Block: 256 thr = 8 waves; block tile 64M x 128N; wave tile 32M x 32N via
// 2x2 register blocking (4 independent WMMA accumulator chains).
// A tile staged via regular loads (needs zero-fill), B tile staged with
// global_load_async_to_lds_b128 + s_wait_asynccnt (CDNA5 async path).
// ---------------------------------------------------------------------------
__global__ __launch_bounds__(256)
void conv_wmma(const _Float16* __restrict__ in, int inH, int inW, int Cin,
               const _Float16* __restrict__ wgt,
               const float* __restrict__ bias, const float* __restrict__ gamma,
               const float* __restrict__ beta, int Co_real, int act,
               _Float16* __restrict__ out, int H, int W, int Cout,
               int ksz, int ups)
{
    __shared__ _Float16 lA[64 * 32];    // 4 KB : 64 rows x 32 K
    __shared__ _Float16 lB[128 * 32];   // 8 KB : 128 n x 32 K (fragment order)

    const int HW   = H * W;
    const int b    = blockIdx.z;
    const int m0   = blockIdx.x * 64;
    const int nblk = blockIdx.y * 128;
    const int tid  = threadIdx.x;
    const int wave = tid >> 5;
    const int lane = tid & 31;
    const int wm   = wave >> 2;          // 0..1  (M sub-block of 32 rows)
    const int wn   = wave & 3;           // 0..3  (N sub-block of 32 cols)
    const int lrow = lane & 15;
    const int half = lane >> 4;
    const int KB   = Cin >> 5;
    const int pad  = ksz >> 1;
    const size_t inImg = (size_t)b * inH * inW;

    // --- staging roles -----------------------------------------------------
    // A: thread -> (row 0..63, 8-channel chunk 0..3)
    const int  sr  = tid >> 2;
    const int  sc  = (tid & 3) << 3;
    const int  sm  = m0 + sr;
    const int  sy  = sm / W;
    const int  sx  = sm % W;
    const bool smv = sm < HW;
    // B: thread -> (n' 0..127, half 0..1), one 16B fragment slice
    const int  bn  = tid >> 1;
    const int  bh  = tid & 1;
    const bool bnv = (nblk + bn) < Cout;
    const unsigned ldsBoff = (unsigned)(size_t)(&lB[(bn * 2 + bh) * 16]);

    // --- wave compute coords -----------------------------------------------
    const int  n0   = nblk + wn * 32 + lrow;
    const int  n1   = n0 + 16;
    const bool nok0 = (nblk + wn * 32) < Cout;
    const bool nok1 = (nblk + wn * 32 + 16) < Cout;

    v8f acc00 = {0.f,0.f,0.f,0.f,0.f,0.f,0.f,0.f};
    v8f acc01 = acc00, acc10 = acc00, acc11 = acc00;

    const size_t bstride = (size_t)Cout * 32;   // per-kb element stride in wgt

    for (int t = 0; t < ksz * ksz; ++t) {
        int iy = sy + (t / ksz) - pad;
        int ix = sx + (t % ksz) - pad;
        bool inb = smv && iy >= 0 && iy < H && ix >= 0 && ix < W;
        int siy = ups ? (iy >> 1) : iy;
        int six = ups ? (ix >> 1) : ix;
        if (!inb) { siy = 0; six = 0; }
        const _Float16* ag = in + (inImg + (size_t)siy * inW + six) * Cin + sc;
        const _Float16* bg = wgt +
            ((((size_t)t * KB) * Cout + nblk + bn) * 2 + bh) * 16;

        for (int kb = 0; kb < KB; ++kb) {
            __syncthreads();                     // previous consume done
            // stage A (zero-filled SAME padding)
            h8 av = {};
            if (inb) av = *(const h8*)(ag + (kb << 5));
            *(h8*)(&lA[sr * 32 + sc]) = av;
            // stage B: async global -> LDS 128-bit copy (ASYNCcnt tracked)
            if (bnv) {
                const _Float16* gsrc = bg + (size_t)kb * bstride;
                asm volatile("global_load_async_to_lds_b128 %0, %1, off"
                             :: "v"(ldsBoff),
                                "v"((unsigned long long)(size_t)gsrc)
                             : "memory");
            }
            asm volatile("s_wait_asynccnt 0x0" ::: "memory");
            __syncthreads();

            // consume: read fragments from LDS
            const _Float16* a0 = &lA[(wm * 32 + lrow) * 32 + half * 8];
            const _Float16* a1 = a0 + 16 * 32;
            h8 a0lo = *(const h8*)(a0);
            h8 a0hi = *(const h8*)(a0 + 16);
            h8 a1lo = *(const h8*)(a1);
            h8 a1hi = *(const h8*)(a1 + 16);
            v16h A0 = __builtin_shufflevector(a0lo, a0hi, 0,1,2,3,4,5,6,7,
                                              8,9,10,11,12,13,14,15);
            v16h A1 = __builtin_shufflevector(a1lo, a1hi, 0,1,2,3,4,5,6,7,
                                              8,9,10,11,12,13,14,15);
            if (nok0) {
                v16h B0 = *(const v16h*)(&lB[((wn * 32 + lrow) * 2 + half) * 16]);
                acc00 = __builtin_amdgcn_wmma_f32_16x16x32_f16(
                            false, A0, false, B0, (short)0, acc00, false, false);
                acc10 = __builtin_amdgcn_wmma_f32_16x16x32_f16(
                            false, A1, false, B0, (short)0, acc10, false, false);
            }
            if (nok1) {
                v16h B1 = *(const v16h*)(&lB[((wn * 32 + 16 + lrow) * 2 + half) * 16]);
                acc01 = __builtin_amdgcn_wmma_f32_16x16x32_f16(
                            false, A0, false, B1, (short)0, acc01, false, false);
                acc11 = __builtin_amdgcn_wmma_f32_16x16x32_f16(
                            false, A1, false, B1, (short)0, acc11, false, false);
            }
        }
    }

    // --- epilogue: D lane layout => row = v + half*8, col = lrow -----------
    float b0 = 0.f, g0 = 1.f, be0 = 0.f;
    float b1 = 0.f, g1 = 1.f, be1 = 0.f;
    if (nok0 && n0 < Co_real) {
        b0 = bias[n0];
        if (act) { g0 = gamma[n0]; be0 = beta[n0]; }
    }
    if (nok1 && n1 < Co_real) {
        b1 = bias[n1];
        if (act) { g1 = gamma[n1]; be1 = beta[n1]; }
    }
#pragma unroll
    for (int s = 0; s < 2; ++s) {
        v8f aN0 = s ? acc10 : acc00;
        v8f aN1 = s ? acc11 : acc01;
#pragma unroll
        for (int v = 0; v < 8; ++v) {
            int mm = m0 + wm * 32 + s * 16 + v + half * 8;
            if (mm < HW) {
                size_t rowoff = ((size_t)b * HW + mm) * Cout;
                if (nok0) {
                    float val = aN0[v] + b0;
                    if (act) { val = g0 * val + be0; val = fmaxf(val, 0.f); }
                    out[rowoff + n0] = (_Float16)val;
                }
                if (nok1) {
                    float val = aN1[v] + b1;
                    if (act) { val = g1 * val + be1; val = fmaxf(val, 0.f); }
                    out[rowoff + n1] = (_Float16)val;
                }
            }
        }
    }
}

// ---------------------------------------------------------------------------
// 2x2 average pool, f16, vectorized over 8 channels per thread.
// H,W are OUTPUT dims (input is 2H x 2W).
// ---------------------------------------------------------------------------
__global__ void avgpool2(const _Float16* __restrict__ in, _Float16* __restrict__ out,
                         int H, int W, int C, int total)
{
    int idx = blockIdx.x * blockDim.x + threadIdx.x;
    if (idx >= total) return;
    int c8   = idx % (C >> 3);
    int p    = idx / (C >> 3);
    int x    = p % W;
    int rest = p / W;
    int y    = rest % H;
    int b    = rest / H;
    const size_t rowStride = (size_t)(W * 2) * C;
    const _Float16* base = in + (((size_t)b * (H * 2) + y * 2) * (W * 2) + x * 2) * C + c8 * 8;
    h8 a  = *(const h8*)(base);
    h8 c1 = *(const h8*)(base + C);
    h8 c2 = *(const h8*)(base + rowStride);
    h8 c3 = *(const h8*)(base + rowStride + C);
    h8 s = a + c1 + c2 + c3;
    h8 r;
#pragma unroll
    for (int i = 0; i < 8; ++i) r[i] = s[i] * (_Float16)0.25f;
    *(h8*)(out + (((size_t)b * H + y) * W + x) * C + c8 * 8) = r;
}

// ---------------------------------------------------------------------------
// Channel concat: dst[0..C1) = s1, dst[C1..C1+C2copy) = s2 (stride Cs2), rest 0
// ---------------------------------------------------------------------------
__global__ void concat_k(_Float16* __restrict__ dst, int Cd,
                         const _Float16* __restrict__ s1, int C1,
                         const _Float16* __restrict__ s2, int Cs2, int C2copy,
                         int total)
{
    int idx = blockIdx.x * blockDim.x + threadIdx.x;
    if (idx >= total) return;
    int c = idx % Cd;
    int p = idx / Cd;
    _Float16 v = (_Float16)0.0f;
    if (c < C1)               v = s1[(size_t)p * C1 + c];
    else if (c - C1 < C2copy) v = s2[(size_t)p * Cs2 + (c - C1)];
    dst[idx] = v;
}

// ---------------------------------------------------------------------------
// Bilinear sample of 3-channel 128x128 f32 image at normalized coords
// ---------------------------------------------------------------------------
__device__ __forceinline__ void bilinear3(const float* __restrict__ img,
                                          float xn, float yn, float* r)
{
    const int Hh = IMG_H, Ww = IMG_W;
    float xf = 0.5f * (xn + 1.0f) * (float)(Ww - 1);
    float yf = 0.5f * (yn + 1.0f) * (float)(Hh - 1);
    float x0f = floorf(xf), y0f = floorf(yf);
    int x0 = (int)x0f, y0 = (int)y0f;
    int x1 = x0 + 1, y1 = y0 + 1;
    int x0c = min(max(x0, 0), Ww - 1), x1c = min(max(x1, 0), Ww - 1);
    int y0c = min(max(y0, 0), Hh - 1), y1c = min(max(y1, 0), Hh - 1);
    float fx0 = (float)x0c, fx1 = (float)x1c;
    float fy0 = (float)y0c, fy1 = (float)y1c;
    float wa = (fx1 - xf) * (fy1 - yf);
    float wb = (fx1 - xf) * (yf - fy0);
    float wc = (xf - fx0) * (fy1 - yf);
    float wd = (xf - fx0) * (yf - fy0);
    const float* Ia = img + ((size_t)y0c * Ww + x0c) * 3;
    const float* Ib = img + ((size_t)y1c * Ww + x0c) * 3;
    const float* Ic = img + ((size_t)y0c * Ww + x1c) * 3;
    const float* Id = img + ((size_t)y1c * Ww + x1c) * 3;
#pragma unroll
    for (int c = 0; c < 3; ++c)
        r[c] = wa * Ia[c] + wb * Ib[c] + wc * Ic[c] + wd * Id[c];
}

// ---------------------------------------------------------------------------
// Build hourglass input (B,H,W,64 f16; real 44ch) + sparse motions (B,H,W,11,2 f32)
// Replicates the reference's tile() batch-wrap: deformed (b,kp) samples
// source[(b*11+kp)%8].
// ---------------------------------------------------------------------------
__global__ void build_input(const float* __restrict__ src,
                            const float* __restrict__ kpdv, const float* __restrict__ kpdj,
                            const float* __restrict__ kpsv, const float* __restrict__ kpsj,
                            _Float16* __restrict__ inp, float* __restrict__ sparse)
{
    int idx = blockIdx.x * blockDim.x + threadIdx.x;
    const int HW = IMG_H * IMG_W;
    if (idx >= BS * HW) return;
    int p = idx % HW;
    int b = idx / HW;
    int x = p % IMG_W;
    int y = p / IMG_W;
    float gx = 2.0f * (float)x / (float)(IMG_W - 1) - 1.0f;
    float gy = 2.0f * (float)y / (float)(IMG_H - 1) - 1.0f;

    _Float16* op = inp + (size_t)idx * 64;
    float*    sp = sparse + (size_t)idx * 22;

    {   // kp = 0 : identity motion, zero heat
        float r[3];
        const float* img = src + (size_t)((b * (NKP + 1)) % BS) * HW * 3;
        bilinear3(img, gx, gy, r);
        op[0] = (_Float16)0.0f;
        op[1] = (_Float16)r[0];
        op[2] = (_Float16)r[1];
        op[3] = (_Float16)r[2];
        sp[0] = gx; sp[1] = gy;
    }
    for (int kp = 1; kp <= NKP; ++kp) {
        int j = kp - 1;
        const float* dv = kpdv + ((size_t)b * NKP + j) * 2;
        const float* sv = kpsv + ((size_t)b * NKP + j) * 2;
        const float* Dj = kpdj + ((size_t)b * NKP + j) * 4;
        const float* Sj = kpsj + ((size_t)b * NKP + j) * 4;
        float ddx = gx - dv[0], ddy = gy - dv[1];
        float dsx = gx - sv[0], dsy = gy - sv[1];
        float heat = __expf(-50.0f * (ddx * ddx + ddy * ddy))
                   - __expf(-50.0f * (dsx * dsx + dsy * dsy)); // 0.5/0.01 = 50
        // jac = Sj @ Dj
        float j00 = Sj[0] * Dj[0] + Sj[1] * Dj[2];
        float j01 = Sj[0] * Dj[1] + Sj[1] * Dj[3];
        float j10 = Sj[2] * Dj[0] + Sj[3] * Dj[2];
        float j11 = Sj[2] * Dj[1] + Sj[3] * Dj[3];
        float sxn = j00 * ddx + j01 * ddy + sv[0];
        float syn = j10 * ddx + j11 * ddy + sv[1];
        float r[3];
        const float* img = src + (size_t)((b * (NKP + 1) + kp) % BS) * HW * 3;
        bilinear3(img, sxn, syn, r);
        op[kp * 4 + 0] = (_Float16)heat;
        op[kp * 4 + 1] = (_Float16)r[0];
        op[kp * 4 + 2] = (_Float16)r[1];
        op[kp * 4 + 3] = (_Float16)r[2];
        sp[kp * 2 + 0] = sxn;
        sp[kp * 2 + 1] = syn;
    }
    for (int c = 44; c < 64; ++c) op[c] = (_Float16)0.0f;
}

// ---------------------------------------------------------------------------
// softmax(mask) -> deformation = sum(mask * sparse); occlusion = sigmoid
// d_out: [0 .. 2N) deformation, [2N .. 3N) occlusion
// ---------------------------------------------------------------------------
__global__ void finalize_k(const _Float16* __restrict__ maskbuf,
                           const _Float16* __restrict__ occbuf,
                           const float* __restrict__ sparse,
                           float* __restrict__ outp)
{
    const int N = BS * IMG_H * IMG_W;
    int p = blockIdx.x * blockDim.x + threadIdx.x;
    if (p >= N) return;
    float m[NKP + 1];
    float mx = -1e30f;
#pragma unroll
    for (int k = 0; k <= NKP; ++k) {
        m[k] = (float)maskbuf[(size_t)p * 16 + k];
        mx = fmaxf(mx, m[k]);
    }
    float s = 0.0f;
#pragma unroll
    for (int k = 0; k <= NKP; ++k) { m[k] = __expf(m[k] - mx); s += m[k]; }
    float inv = 1.0f / s;
    float dx = 0.0f, dy = 0.0f;
#pragma unroll
    for (int k = 0; k <= NKP; ++k) {
        float w = m[k] * inv;
        dx += w * sparse[(size_t)p * 22 + 2 * k];
        dy += w * sparse[(size_t)p * 22 + 2 * k + 1];
    }
    outp[(size_t)p * 2 + 0] = dx;
    outp[(size_t)p * 2 + 1] = dy;
    float o = (float)occbuf[(size_t)p * 16];
    outp[(size_t)N * 2 + p] = 1.0f / (1.0f + __expf(-o));
}

// ===========================================================================
// Host launch
// ===========================================================================
static inline size_t align256(size_t x) { return (x + 255) & ~(size_t)255; }

extern "C" void kernel_launch(void* const* d_in, const int* in_sizes, int n_in,
                              void* d_out, int out_size, void* d_ws, size_t ws_size,
                              hipStream_t stream)
{
    (void)n_in; (void)out_size; (void)ws_size;

    // ---- resolve input pointers (handles insertion-order or sorted pytree) --
    const float *src, *kpdv, *kpdj, *kpsv, *kpsj, *maskw, *maskb, *occw, *occb;
    const float *ew[5], *ebb[5], *eg[5], *ebe[5];
    const float *dw[5], *db[5], *dg[5], *dbe[5];

    if (in_sizes[0] == BS * IMG_H * IMG_W * 3) {
        // dict insertion order; blocks as (w, b, gamma, beta)
        src  = (const float*)d_in[0];
        kpdv = (const float*)d_in[1];
        kpdj = (const float*)d_in[2];
        kpsv = (const float*)d_in[3];
        kpsj = (const float*)d_in[4];
        int p = 5;
        for (int i = 0; i < 5; ++i) {
            ew[i]  = (const float*)d_in[p + 0];
            ebb[i] = (const float*)d_in[p + 1];
            eg[i]  = (const float*)d_in[p + 2];
            ebe[i] = (const float*)d_in[p + 3];
            p += 4;
        }
        for (int i = 0; i < 5; ++i) {
            dw[i]  = (const float*)d_in[p + 0];
            db[i]  = (const float*)d_in[p + 1];
            dg[i]  = (const float*)d_in[p + 2];
            dbe[i] = (const float*)d_in[p + 3];
            p += 4;
        }
        maskw = (const float*)d_in[45];
        maskb = (const float*)d_in[46];
        occw  = (const float*)d_in[47];
        occb  = (const float*)d_in[48];
    } else {
        // jax tree_flatten: dict keys sorted; blocks as (b, beta, gamma, w)
        kpdj = (const float*)d_in[0];
        kpdv = (const float*)d_in[1];
        kpsj = (const float*)d_in[2];
        kpsv = (const float*)d_in[3];
        int p = 4;
        for (int i = 0; i < 5; ++i) {
            db[i]  = (const float*)d_in[p + 0];
            dbe[i] = (const float*)d_in[p + 1];
            dg[i]  = (const float*)d_in[p + 2];
            dw[i]  = (const float*)d_in[p + 3];
            p += 4;
        }
        for (int i = 0; i < 5; ++i) {
            ebb[i] = (const float*)d_in[p + 0];
            ebe[i] = (const float*)d_in[p + 1];
            eg[i]  = (const float*)d_in[p + 2];
            ew[i]  = (const float*)d_in[p + 3];
            p += 4;
        }
        maskb = (const float*)d_in[44];
        maskw = (const float*)d_in[45];
        occb  = (const float*)d_in[46];
        occw  = (const float*)d_in[47];
        src   = (const float*)d_in[48];
    }

    // ---- workspace arena ----------------------------------------------------
    char* ws = (char*)d_ws;
    size_t off = 0;
    auto alloc = [&](size_t bytes) -> char* {
        char* pt = ws + off;
        off = align256(off + bytes);
        return pt;
    };

    // layer specs
    const int encCiR[5] = {44, 128, 256, 512, 1024};
    const int encCiP[5] = {64, 128, 256, 512, 1024};
    const int encCo[5]  = {128, 256, 512, 1024, 1024};
    const int decCi[5]  = {1024, 2048, 1024, 512, 256};
    const int decCo[5]  = {1024, 512, 256, 128, 64};

    _Float16 *WE[5], *WD[5], *WM, *WO;
    for (int i = 0; i < 5; ++i)
        WE[i] = (_Float16*)alloc((size_t)9 * encCiP[i] * encCo[i] * 2);
    for (int i = 0; i < 5; ++i)
        WD[i] = (_Float16*)alloc((size_t)9 * decCi[i] * decCo[i] * 2);
    WM = (_Float16*)alloc((size_t)49 * 128 * 16 * 2);
    WO = (_Float16*)alloc((size_t)49 * 128 * 16 * 2);

    const size_t HW = (size_t)IMG_H * IMG_W;
    _Float16* A_IN = (_Float16*)alloc(BS * HW * 64 * 2);
    _Float16* S0 = (_Float16*)alloc((size_t)BS * 64 * 64 * 128 * 2);
    _Float16* S1 = (_Float16*)alloc((size_t)BS * 32 * 32 * 256 * 2);
    _Float16* S2 = (_Float16*)alloc((size_t)BS * 16 * 16 * 512 * 2);
    _Float16* S3 = (_Float16*)alloc((size_t)BS * 8 * 8 * 1024 * 2);
    _Float16* S4 = (_Float16*)alloc((size_t)BS * 4 * 4 * 1024 * 2);
    _Float16* T0 = (_Float16*)alloc(BS * HW * 128 * 2);
    _Float16* T1 = (_Float16*)alloc(BS * HW * 128 * 2);
    _Float16* MASKB = (_Float16*)alloc(BS * HW * 16 * 2);
    _Float16* OCCB  = (_Float16*)alloc(BS * HW * 16 * 2);
    float* SPARSE = (float*)alloc(BS * HW * 22 * 4);

    // ---- weight repack ------------------------------------------------------
    auto cw = [&](const float* w, int taps, int cir, int cip, int cor, int cop,
                  _Float16* dst) {
        int total = taps * (cip / 32) * cop * 32;
        convert_w<<<(total + 255) / 256, 256, 0, stream>>>(w, taps, cir, cip,
                                                           cor, cop, dst, total);
    };
    for (int i = 0; i < 5; ++i)
        cw(ew[i], 9, encCiR[i], encCiP[i], encCo[i], encCo[i], WE[i]);
    for (int i = 0; i < 5; ++i)
        cw(dw[i], 9, decCi[i], decCi[i], decCo[i], decCo[i], WD[i]);
    cw(maskw, 49, 108, 128, 11, 16, WM);
    cw(occw,  49, 108, 128, 1,  16, WO);

    // ---- build input + sparse motions --------------------------------------
    {
        int total = BS * (int)HW;
        build_input<<<(total + 127) / 128, 128, 0, stream>>>(
            src, kpdv, kpdj, kpsv, kpsj, A_IN, SPARSE);
    }

    auto conv = [&](const _Float16* in, int inH, int inW, int Cin,
                    const _Float16* wgt, const float* bias, const float* gamma,
                    const float* beta, int CoReal, int act, _Float16* out,
                    int H, int W, int Cout, int ksz, int ups) {
        dim3 g((H * W + 63) / 64, (Cout + 127) / 128, BS);
        conv_wmma<<<g, 256, 0, stream>>>(in, inH, inW, Cin, wgt, bias, gamma,
                                         beta, CoReal, act, out, H, W, Cout,
                                         ksz, ups);
    };
    auto pool = [&](const _Float16* in, _Float16* out, int H, int W, int C) {
        int total = BS * H * W * (C / 8);
        avgpool2<<<(total + 255) / 256, 256, 0, stream>>>(in, out, H, W, C, total);
    };
    auto cat = [&](_Float16* dst, int Cd, const _Float16* s1, int C1,
                   const _Float16* s2, int Cs2, int C2copy, int npix) {
        int total = BS * npix * Cd;
        concat_k<<<(total + 255) / 256, 256, 0, stream>>>(dst, Cd, s1, C1, s2,
                                                          Cs2, C2copy, total);
    };

    // ---- encoder ------------------------------------------------------------
    conv(A_IN, 128, 128, 64,   WE[0], ebb[0], eg[0], ebe[0], 128,  1, T0, 128, 128, 128, 3, 0);
    pool(T0, S0, 64, 64, 128);
    conv(S0,   64,  64, 128,  WE[1], ebb[1], eg[1], ebe[1], 256,  1, T0, 64, 64, 256, 3, 0);
    pool(T0, S1, 32, 32, 256);
    conv(S1,   32,  32, 256,  WE[2], ebb[2], eg[2], ebe[2], 512,  1, T0, 32, 32, 512, 3, 0);
    pool(T0, S2, 16, 16, 512);
    conv(S2,   16,  16, 512,  WE[3], ebb[3], eg[3], ebe[3], 1024, 1, T0, 16, 16, 1024, 3, 0);
    pool(T0, S3, 8, 8, 1024);
    conv(S3,    8,   8, 1024, WE[4], ebb[4], eg[4], ebe[4], 1024, 1, T0, 8, 8, 1024, 3, 0);
    pool(T0, S4, 4, 4, 1024);

    // ---- decoder (upsample fused into conv input addressing) ---------------
    conv(S4, 4, 4, 1024,   WD[0], db[0], dg[0], dbe[0], 1024, 1, T0, 8, 8, 1024, 3, 1);
    cat(T1, 2048, T0, 1024, S3, 1024, 1024, 64);
    conv(T1, 8, 8, 2048,   WD[1], db[1], dg[1], dbe[1], 512,  1, T0, 16, 16, 512, 3, 1);
    cat(T1, 1024, T0, 512, S2, 512, 512, 256);
    conv(T1, 16, 16, 1024, WD[2], db[2], dg[2], dbe[2], 256,  1, T0, 32, 32, 256, 3, 1);
    cat(T1, 512, T0, 256, S1, 256, 256, 1024);
    conv(T1, 32, 32, 512,  WD[3], db[3], dg[3], dbe[3], 128,  1, T0, 64, 64, 128, 3, 1);
    cat(T1, 256, T0, 128, S0, 128, 128, 4096);
    conv(T1, 64, 64, 256,  WD[4], db[4], dg[4], dbe[4], 64,   1, T0, 128, 128, 64, 3, 1);
    // final concat with original (44-real-channel) input, pad to 128
    cat(T1, 128, T0, 64, A_IN, 64, 44, 16384);

    // ---- heads (7x7) --------------------------------------------------------
    conv(T1, 128, 128, 128, WM, maskb, nullptr, nullptr, 11, 0, MASKB, 128, 128, 16, 7, 0);
    conv(T1, 128, 128, 128, WO, occb,  nullptr, nullptr, 1,  0, OCCB,  128, 128, 16, 7, 0);

    // ---- softmax + deformation + occlusion ----------------------------------
    {
        int total = BS * (int)HW;
        finalize_k<<<(total + 255) / 256, 256, 0, stream>>>(
            MASKB, OCCB, SPARSE, (float*)d_out);
    }
}
